// LinGNN_87162066305784
// MI455X (gfx1250) — matrix operand order repaired and checked by reading, not compiled, not run on hardware
//
#include <hip/hip_runtime.h>
#include <hip/hip_bf16.h>
#include <stdint.h>

// ---------------- problem constants ----------------
#define NN 20000          // nodes
#define NE 320000         // edges
#define MPAD 20096        // nodes padded to multiple of 128 for GEMM row blocking
#define EPSBN 1e-5f
#define BSTR 40           // LDS row stride (elements) for B tile: conflict-free, 16B aligned

typedef __bf16 bf8v  __attribute__((ext_vector_type(8)));
typedef __bf16 bf16v __attribute__((ext_vector_type(16)));
typedef float  f8v   __attribute__((ext_vector_type(8)));

__device__ __forceinline__ float lrelu(float v, float s) { return v > 0.f ? v : s * v; }

// ---- CDNA5 async global->LDS staging (ASYNCcnt path), via inline asm ----
// VDST = 32-bit LDS byte address (flat shared addr low 32 bits), VADDR = 64-bit global addr.
#if defined(__gfx1250__)
#define STAGE_B16(gp, lp)                                                         \
    do {                                                                          \
        unsigned _lo = (unsigned)(size_t)(lp);                                    \
        unsigned long long _ga = (unsigned long long)(size_t)(gp);                \
        asm volatile("global_load_async_to_lds_b128 %0, %1, off"                  \
                     :: "v"(_lo), "v"(_ga) : "memory");                           \
    } while (0)
#define WAIT_STAGE() asm volatile("s_wait_asynccnt 0x0" ::: "memory")
#else
#define STAGE_B16(gp, lp)                                                         \
    do { bf8v _t = *(const bf8v*)(gp); *(bf8v*)(lp) = _t; } while (0)
#define WAIT_STAGE() do {} while (0)
#endif

// ---------------- utility kernels ----------------
__global__ void k_zero_u32(unsigned* p, long long n) {
    long long i = (long long)blockIdx.x * blockDim.x + threadIdx.x;
    if (i < n) p[i] = 0u;
}

// copy x [rows,cs] into padded [rows,cd] fp32 buffer (pad cols = 0)
__global__ void k_pad_copy(const float* __restrict__ src, float* __restrict__ dst,
                           int rows, int cs, int cd) {
    long long i = (long long)blockIdx.x * blockDim.x + threadIdx.x;
    long long n = (long long)rows * cd;
    if (i >= n) return;
    int c = (int)(i % cd);
    int r = (int)(i / cd);
    dst[i] = (c < cs) ? src[(long long)r * cs + c] : 0.f;
}

// ---------------- CSR build (once per launch; graph is static) ----------------
__global__ void k_deg(const int* __restrict__ dst, int* __restrict__ deg, int E) {
    int e = blockIdx.x * blockDim.x + threadIdx.x;
    if (e < E) atomicAdd(&deg[dst[e]], 1);
}

// single-block exclusive scan of deg[0..n-1] -> rowptr[0..n]
__global__ void k_scan(const int* __restrict__ deg, int* __restrict__ rowptr, int n) {
    __shared__ int part[256];
    int tid = threadIdx.x;
    int chunk = (n + 255) / 256;
    int lo = tid * chunk, hi = min(n, lo + chunk);
    int s = 0;
    for (int i = lo; i < hi; ++i) s += deg[i];
    part[tid] = s;
    __syncthreads();
    if (tid == 0) {
        int acc = 0;
        for (int i = 0; i < 256; ++i) { int t = part[i]; part[i] = acc; acc += t; }
    }
    __syncthreads();
    int acc = part[tid];
    for (int i = lo; i < hi; ++i) { rowptr[i] = acc; acc += deg[i]; }
    if (tid == 255) rowptr[n] = acc;
}

__global__ void k_copy_i32(const int* __restrict__ a, int* __restrict__ b, int n) {
    int i = blockIdx.x * blockDim.x + threadIdx.x;
    if (i < n) b[i] = a[i];
}

__global__ void k_fill_csr(const int* __restrict__ dst, int* __restrict__ pos,
                           int* __restrict__ eord, int E) {
    int e = blockIdx.x * blockDim.x + threadIdx.x;
    if (e >= E) return;
    int p = atomicAdd(&pos[dst[e]], 1);
    eord[p] = e;
}

// ---------------- batch-norm (training mode, biased var) ----------------
__global__ void k_colstat(const float* __restrict__ X, int rows, int KS,
                          float* __restrict__ stats) {
    int c = blockIdx.x * 32 + (threadIdx.x & 31);
    int chunk = (rows + gridDim.y - 1) / gridDim.y;
    int rlo = blockIdx.y * chunk + (threadIdx.x >> 5);
    int rhi = min(rows, (int)((blockIdx.y + 1) * chunk));
    float s = 0.f, q = 0.f;
    for (int r = rlo; r < rhi; r += 8) {
        float v = X[(long long)r * KS + c];
        s += v; q += v * v;
    }
    atomicAdd(&stats[c], s);
    atomicAdd(&stats[KS + c], q);
}

__global__ void k_bn_finalize(const float* __restrict__ stats, int KS, int O, float inv,
                              const float* __restrict__ g, const float* __restrict__ b,
                              float* __restrict__ scale, float* __restrict__ shift) {
    int c = blockIdx.x * blockDim.x + threadIdx.x;
    if (c >= KS) return;
    if (c >= O) { scale[c] = 0.f; shift[c] = 0.f; return; }
    float m = stats[c] * inv;
    float v = stats[KS + c] * inv - m * m;
    float sc = g[c] * rsqrtf(v + EPSBN);
    scale[c] = sc;
    shift[c] = b[c] - m * sc;
}

// bn-apply (optional) + fp32->bf16; rows >= rowsValid are zero-filled (GEMM row padding)
__global__ void k_prep_bf16(const float* __restrict__ X, const float* __restrict__ scale,
                            const float* __restrict__ shift, int doAff,
                            __bf16* __restrict__ XB, long long n, int KS, int rowsValid) {
    long long i = (long long)blockIdx.x * blockDim.x + threadIdx.x;
    if (i >= n) return;
    long long r = i / KS;
    float v = (r < rowsValid) ? X[i] : 0.f;
    if (doAff) { int c = (int)(i % KS); v = v * scale[c] + shift[c]; }
    XB[i] = (__bf16)v;
}

// weight [O,K] fp32 -> padded bf16 [OP,KWS]
__global__ void k_wconv(const float* __restrict__ W, int O, int K,
                        __bf16* __restrict__ WB, int OP, int KWS) {
    long long i = (long long)blockIdx.x * blockDim.x + threadIdx.x;
    long long n = (long long)OP * KWS;
    if (i >= n) return;
    int k = (int)(i % KWS);
    int o = (int)(i / KWS);
    WB[i] = (o < O && k < K) ? (__bf16)W[(long long)o * K + k] : (__bf16)0.f;
}

// ---------------- WMMA bf16 GEMM: Y[M,OS] = XB[M,KS] * WB^T + bias ----------------
// 8 waves/block; block = 128 rows x 64 cols; B tile (64x32) staged in LDS, double-buffered,
// via async global->LDS copies tracked on ASYNCcnt.
// grid = (MPAD/128, OP/64)
__global__ __launch_bounds__(256)
void k_gemm_wmma(const __bf16* __restrict__ XB, int KS,
                 const __bf16* __restrict__ WB, int KWS,
                 const float* __restrict__ bias, int O,
                 float* __restrict__ Y, int OS, int kIters, int Mrows) {
    __shared__ __bf16 Bs[2][64 * BSTR];

    const int tid = threadIdx.x;
    const int wv = tid >> 5, l = tid & 31;
    const int rowbase = blockIdx.x * 128 + wv * 16;
    const int colbase = blockIdx.y * 64;

    // cooperative B staging: thread t moves 16B; row = t>>2, 16B segment = t&3
    const int brow = tid >> 2, bseg = tid & 3;
    const __bf16* bsrc = WB + (long long)(colbase + brow) * KWS + bseg * 8;
    __bf16* bdst0 = &Bs[0][0] + brow * BSTR + bseg * 8;
    __bf16* bdst1 = &Bs[1][0] + brow * BSTR + bseg * 8;

    // A addressing (16-bit 16x32 layout): lanes 0-15 -> K 0..7 & 16..23; 16-31 -> 8..15 & 24..31
    const int arow  = rowbase + (l & 15);
    const int akoff = (l < 16) ? 0 : 8;
    const int bkoff = (l < 16) ? 0 : 8;    // B frag: lanes 0-15 K0..7 & 16..23, lanes 16-31 K8..15 & 24..31
    const __bf16* aptr = XB + (long long)arow * KS + akoff;

    f8v acc[4];
#pragma unroll
    for (int t = 0; t < 4; ++t)
#pragma unroll
        for (int r = 0; r < 8; ++r) acc[t][r] = 0.f;

    // stage k=0 into buffer 0
    STAGE_B16(bsrc, bdst0);

    int cur = 0;
    for (int ki = 0; ki < kIters; ++ki) {
        WAIT_STAGE();       // this wave's outstanding async copies done
        __syncthreads();    // all waves staged; previous iteration's LDS reads complete

        if (ki + 1 < kIters) {   // stage next k into the other buffer
            const __bf16* ns = bsrc + (ki + 1) * 32;
            __bf16* nd = (cur == 0) ? bdst1 : bdst0;
            STAGE_B16(ns, nd);
        }

        const int kk = ki * 32;
        bf8v alo = *(const bf8v*)(aptr + kk);
        bf8v ahi = *(const bf8v*)(aptr + kk + 16);
        bf16v a = __builtin_shufflevector(alo, ahi, 0,1,2,3,4,5,6,7,8,9,10,11,12,13,14,15);
#pragma unroll
        for (int t = 0; t < 4; ++t) {
            const __bf16* bls = &Bs[cur][0] + (t * 16 + (l & 15)) * BSTR + bkoff;
            bf8v blo = *(const bf8v*)bls;
            bf8v bhi = *(const bf8v*)(bls + 16);
            bf16v b = __builtin_shufflevector(blo, bhi, 0,1,2,3,4,5,6,7,8,9,10,11,12,13,14,15);
            acc[t] = __builtin_amdgcn_wmma_f32_16x16x32_bf16(
                false, a, false, b, (short)0, acc[t], false, false);
        }
        cur ^= 1;
    }

    // C/D layout: VGPR r, lanes 0-15 -> M=r, lanes 16-31 -> M=8+r; N = lane&15
    const int rbump = (l >= 16) ? 8 : 0;
#pragma unroll
    for (int t = 0; t < 4; ++t) {
        int col = colbase + t * 16 + (l & 15);
        if (col < OS) {
            float bv = (col < O) ? bias[col] : 0.f;
#pragma unroll
            for (int r = 0; r < 8; ++r) {
                int row = rowbase + r + rbump;
                if (row < Mrows)
                    Y[(long long)row * OS + col] = acc[t][r] + bv;
            }
        }
    }
}

// ---------------- edge attr batch norm (2 cols) ----------------
__global__ void k_ea_stat(const float* __restrict__ ea, int E, float* __restrict__ stats) {
    int tid = blockIdx.x * blockDim.x + threadIdx.x;
    int stride = gridDim.x * blockDim.x;
    float s0 = 0, q0 = 0, s1 = 0, q1 = 0;
    for (int e = tid; e < E; e += stride) {
        float a = ea[2 * e], b = ea[2 * e + 1];
        s0 += a; q0 += a * a; s1 += b; q1 += b * b;
    }
    atomicAdd(&stats[0], s0); atomicAdd(&stats[1], s1);
    atomicAdd(&stats[2], q0); atomicAdd(&stats[3], q1);
}

__global__ void k_ea_finalize(const float* __restrict__ stats, float inv,
                              const float* __restrict__ g, const float* __restrict__ b,
                              float* __restrict__ scale, float* __restrict__ shift) {
    int c = threadIdx.x;
    if (c >= 2) return;
    float m = stats[c] * inv;
    float v = stats[2 + c] * inv - m * m;
    float sc = g[c] * rsqrtf(v + EPSBN);
    scale[c] = sc;
    shift[c] = b[c] - m * sc;
}

__global__ void k_ea_apply(const float* __restrict__ ea, const float* __restrict__ scale,
                           const float* __restrict__ shift, float* __restrict__ out, long long n) {
    long long i = (long long)blockIdx.x * blockDim.x + threadIdx.x;
    if (i >= n) return;
    int c = (int)(i & 1);
    out[i] = ea[i] * scale[c] + shift[c];
}

// ---------------- edge phase (atomic-free via CSR) ----------------
// one wave per edge: s_e = sum_c lrelu(xl[src,c]+xr[dst,c]+ea0*We[c,0]+ea1*We[c,1], .2)*att[c]
__global__ void k_edge_logits(const float* __restrict__ XL, const float* __restrict__ XR,
                              int HS, int O, const float* __restrict__ EAn,
                              const int* __restrict__ src, const int* __restrict__ dst,
                              const float* __restrict__ We, const float* __restrict__ att,
                              float* __restrict__ S, int E) {
    int e = blockIdx.x * 8 + (threadIdx.x >> 5);
    if (e >= E) return;
    int l = threadIdx.x & 31;
    int sn = src[e], dn = dst[e];
    float ea0 = EAn[2 * e], ea1 = EAn[2 * e + 1];
    const float* xl = XL + (long long)sn * HS;
    const float* xr = XR + (long long)dn * HS;
    float acc = 0.f;
    for (int c = l; c < O; c += 32) {
        float v = xl[c] + xr[c] + ea0 * We[2 * c] + ea1 * We[2 * c + 1];
        acc += lrelu(v, 0.2f) * att[c];
    }
#pragma unroll
    for (int m = 16; m > 0; m >>= 1) acc += __shfl_xor(acc, m, 32);
    if (l == 0) S[e] = acc;
}

// one wave per node: segment max + sum(exp) over its CSR edge list
__global__ void k_node_softmax(const int* __restrict__ rowptr, const int* __restrict__ eord,
                               const float* __restrict__ S, float* __restrict__ SM,
                               float* __restrict__ DEN, int Nn) {
    int n = blockIdx.x * 8 + (threadIdx.x >> 5);
    if (n >= Nn) return;
    int l = threadIdx.x & 31;
    int a = rowptr[n], b = rowptr[n + 1];
    float m = -__builtin_inff();
    for (int j = a + l; j < b; j += 32) m = fmaxf(m, S[eord[j]]);
#pragma unroll
    for (int k = 16; k > 0; k >>= 1) m = fmaxf(m, __shfl_xor(m, k, 32));
    float d = 0.f;
    for (int j = a + l; j < b; j += 32) d += __expf(S[eord[j]] - m);
#pragma unroll
    for (int k = 16; k > 0; k >>= 1) d += __shfl_xor(d, k, 32);
    if (l == 0) { SM[n] = m; DEN[n] = d; }
}

// alpha_e = exp(s_e - smax[dst]) / den[dst]   (in place into S)
__global__ void k_alpha(float* __restrict__ S, const int* __restrict__ dst,
                        const float* __restrict__ SM, const float* __restrict__ DEN, int E) {
    int e = blockIdx.x * blockDim.x + threadIdx.x;
    if (e >= E) return;
    int d = dst[e];
    S[e] = __expf(S[e] - SM[d]) / DEN[d];
}

// one wave per node: OUT[n,:] = bias + sum_e alpha_e * xl[src_e,:]  (no atomics)
__global__ void k_aggregate(const int* __restrict__ rowptr, const int* __restrict__ eord,
                            const int* __restrict__ src, const float* __restrict__ A,
                            const float* __restrict__ XL, int HS, int O,
                            const float* __restrict__ bias, float* __restrict__ OUT, int Nn) {
    int n = blockIdx.x * 8 + (threadIdx.x >> 5);
    if (n >= Nn) return;
    int l = threadIdx.x & 31;
    int a = rowptr[n], b = rowptr[n + 1];
    float* o = OUT + (long long)n * HS;
    for (int c = l; c < HS; c += 32) {
        float acc = 0.f;
        for (int j = a; j < b; ++j) {          // j, e, src[e], A[e] are wave-uniform -> s-loads
            int e = eord[j];
            acc += A[e] * XL[(long long)src[e] * HS + c];
        }
        o[c] = acc + ((c < O) ? bias[c] : 0.f);
    }
}

// mode 0: X=lrelu(OUT); 1: X=lrelu(OUT)+RES; 2: X=OUT
__global__ void k_act(const float* __restrict__ OUT, const float* __restrict__ RES,
                      float* __restrict__ X, long long n, int mode) {
    long long i = (long long)blockIdx.x * blockDim.x + threadIdx.x;
    if (i >= n) return;
    float v = OUT[i];
    if (mode == 0)      X[i] = lrelu(v, 0.01f);
    else if (mode == 1) X[i] = lrelu(v, 0.01f) + RES[i];
    else                X[i] = v;
}

// final [800 -> 1] head: one wave per row
__global__ void k_final_dot(const float* __restrict__ X, const float* __restrict__ W,
                            const float* __restrict__ b, float* __restrict__ out,
                            int rows, int K) {
    int row = blockIdx.x * 8 + (threadIdx.x >> 5);
    if (row >= rows) return;
    int l = threadIdx.x & 31;
    const float* x = X + (long long)row * K;
    float acc = 0.f;
    for (int c = l; c < K; c += 32) acc += x[c] * W[c];
#pragma unroll
    for (int m = 16; m > 0; m >>= 1) acc += __shfl_xor(acc, m, 32);
    if (l == 0) out[row] = acc + b[0];
}

// ---------------- host orchestration ----------------
extern "C" void kernel_launch(void* const* d_in, const int* in_sizes, int n_in,
                              void* d_out, int out_size, void* d_ws, size_t ws_size,
                              hipStream_t stream) {
    (void)in_sizes; (void)n_in; (void)out_size; (void)ws_size;
    const float* x_in  = (const float*)d_in[0];
    const float* ea_in = (const float*)d_in[1];
    const int*   ei    = (const int*)d_in[2];
    const int* src = ei;
    const int* dst = ei + NE;
    auto Wl    = [&](int i) { return (const float*)d_in[7 + i]; };
    auto bl    = [&](int i) { return (const float*)d_in[27 + i]; };
    auto Wr    = [&](int i) { return (const float*)d_in[47 + i]; };
    auto brr   = [&](int i) { return (const float*)d_in[67 + i]; };
    auto We    = [&](int i) { return (const float*)d_in[87 + i]; };
    auto att   = [&](int i) { return (const float*)d_in[107 + i]; };
    auto cbias = [&](int i) { return (const float*)d_in[127 + i]; };
    auto bng   = [&](int i) { return (const float*)d_in[147 + i]; };
    auto bnb   = [&](int i) { return (const float*)d_in[165 + i]; };
    auto linW  = [&](int i) { return (const float*)d_in[183 + i]; };
    auto linb  = [&](int i) { return (const float*)d_in[189 + i]; };

    // workspace carve-out
    char* base = (char*)d_ws;
    size_t off = 0;
    auto carve = [&](size_t bytes) {
        void* p = base + off;
        off = (off + bytes + 255) & ~(size_t)255;
        return p;
    };
    float*  X     = (float*)carve((size_t)NN * 800 * 4);
    float*  OUT   = (float*)carve((size_t)NN * 800 * 4);
    float*  XL    = (float*)carve((size_t)NN * 800 * 4);
    float*  XR    = (float*)carve((size_t)NN * 800 * 4);
    __bf16* XB    = (__bf16*)carve((size_t)MPAD * 800 * 2);
    __bf16* WLB   = (__bf16*)carve((size_t)832 * 800 * 2);
    __bf16* WRB   = (__bf16*)carve((size_t)832 * 800 * 2);
    float*  EAn   = (float*)carve((size_t)NE * 2 * 4);
    float*  S     = (float*)carve((size_t)NE * 4);
    float*  SM    = (float*)carve((size_t)NN * 4);
    float*  DEN   = (float*)carve((size_t)NN * 4);
    float*  STATS = (float*)carve((size_t)1600 * 4);
    float*  SCALE = (float*)carve((size_t)800 * 4);
    float*  SHIFT = (float*)carve((size_t)800 * 4);
    int*    DEG   = (int*)carve((size_t)NN * 4);
    int*    ROWP  = (int*)carve((size_t)(NN + 1) * 4);
    int*    POS   = (int*)carve((size_t)NN * 4);
    int*    EORD  = (int*)carve((size_t)NE * 4);

    auto gsz = [](long long n, int b) { return (unsigned)((n + b - 1) / b); };

    // -------- CSR build (graph is static across layers) --------
    k_zero_u32<<<gsz(NN, 256), 256, 0, stream>>>((unsigned*)DEG, NN);
    k_deg<<<gsz(NE, 256), 256, 0, stream>>>(dst, DEG, NE);
    k_scan<<<1, 256, 0, stream>>>(DEG, ROWP, NN);
    k_copy_i32<<<gsz(NN, 256), 256, 0, stream>>>(ROWP, POS, NN);
    k_fill_csr<<<gsz(NE, 256), 256, 0, stream>>>(dst, POS, EORD, NE);

    // -------- input batchnorms --------
    k_pad_copy<<<gsz((long long)NN * 32, 256), 256, 0, stream>>>(x_in, X, NN, 3, 32);
    k_zero_u32<<<1, 8, 0, stream>>>((unsigned*)STATS, 8);
    k_ea_stat<<<128, 256, 0, stream>>>(ea_in, NE, STATS);
    k_ea_finalize<<<1, 32, 0, stream>>>(STATS, 1.f / NE, (const float*)d_in[5],
                                        (const float*)d_in[6], SCALE, SHIFT);
    k_ea_apply<<<gsz((long long)NE * 2, 256), 256, 0, stream>>>(ea_in, SCALE, SHIFT, EAn,
                                                                (long long)NE * 2);

    // -------- GATv2 stack --------
    int KS = 32, Kin = 3;
    for (int li = 0; li < 20; ++li) {
        const int O  = (li == 19) ? 800 : 700;
        const int OS = (li == 19) ? 800 : 704;
        const int gy = (li == 19) ? 13 : 11;
        const int OP = gy * 64;
        const int kIters = KS / 32;

        if (li <= 18) {
            const float* g = (li == 0) ? (const float*)d_in[3] : bng(li - 1);
            const float* b = (li == 0) ? (const float*)d_in[4] : bnb(li - 1);
            int Ocols = (li == 0) ? 3 : 700;
            k_zero_u32<<<gsz(2 * KS, 256), 256, 0, stream>>>((unsigned*)STATS, 2 * KS);
            k_colstat<<<dim3(KS / 32, 32), 256, 0, stream>>>(X, NN, KS, STATS);
            k_bn_finalize<<<gsz(KS, 256), 256, 0, stream>>>(STATS, KS, Ocols, 1.f / NN,
                                                            g, b, SCALE, SHIFT);
            k_prep_bf16<<<gsz((long long)MPAD * KS, 256), 256, 0, stream>>>(
                X, SCALE, SHIFT, 1, XB, (long long)MPAD * KS, KS, NN);
        } else {
            k_prep_bf16<<<gsz((long long)MPAD * KS, 256), 256, 0, stream>>>(
                X, SCALE, SHIFT, 0, XB, (long long)MPAD * KS, KS, NN);
        }

        k_wconv<<<gsz((long long)OP * KS, 256), 256, 0, stream>>>(Wl(li), O, Kin, WLB, OP, KS);
        k_wconv<<<gsz((long long)OP * KS, 256), 256, 0, stream>>>(Wr(li), O, Kin, WRB, OP, KS);
        k_gemm_wmma<<<dim3(MPAD / 128, gy), 256, 0, stream>>>(XB, KS, WLB, KS, bl(li), O,
                                                              XL, OS, kIters, NN);
        k_gemm_wmma<<<dim3(MPAD / 128, gy), 256, 0, stream>>>(XB, KS, WRB, KS, brr(li), O,
                                                              XR, OS, kIters, NN);

        k_edge_logits<<<gsz(NE, 8), 256, 0, stream>>>(XL, XR, OS, O, EAn, src, dst,
                                                      We(li), att(li), S, NE);
        k_node_softmax<<<gsz(NN, 8), 256, 0, stream>>>(ROWP, EORD, S, SM, DEN, NN);
        k_alpha<<<gsz(NE, 256), 256, 0, stream>>>(S, dst, SM, DEN, NE);
        k_aggregate<<<gsz(NN, 8), 256, 0, stream>>>(ROWP, EORD, src, S, XL, OS, O,
                                                    cbias(li), OUT, NN);

        int mode = (li == 0) ? 0 : (li == 19 ? 2 : 1);
        k_act<<<gsz((long long)NN * OS, 256), 256, 0, stream>>>(OUT, X, X, (long long)NN * OS, mode);

        KS = OS;
        Kin = O;
    }

    // -------- residual linear stack (800 -> 800) x5 --------
    for (int i = 0; i < 5; ++i) {
        k_prep_bf16<<<gsz((long long)MPAD * 800, 256), 256, 0, stream>>>(
            X, SCALE, SHIFT, 0, XB, (long long)MPAD * 800, 800, NN);
        k_wconv<<<gsz((long long)832 * 800, 256), 256, 0, stream>>>(linW(i), 800, 800, WLB, 832, 800);
        k_gemm_wmma<<<dim3(MPAD / 128, 13), 256, 0, stream>>>(XB, 800, WLB, 800, linb(i), 800,
                                                              OUT, 800, 25, NN);
        k_act<<<gsz((long long)NN * 800, 256), 256, 0, stream>>>(OUT, X, X, (long long)NN * 800, 1);
    }

    // -------- final head [800 -> 1] --------
    k_final_dot<<<gsz(NN, 8), 256, 0, stream>>>(X, linW(5), linb(5), (float*)d_out, NN, 800);
}
